// AdaptiveGraphLearner_46866683134384
// MI455X (gfx1250) — matrix operand order repaired and checked by reading, not compiled, not run on hardware
//
#include <hip/hip_runtime.h>

typedef __attribute__((ext_vector_type(16))) _Float16 v16h;
typedef __attribute__((ext_vector_type(8)))  _Float16 v8h;
typedef __attribute__((ext_vector_type(8)))  float    v8f;
typedef __attribute__((ext_vector_type(4)))  float    v4f;

#define NB 16
#define NN 2048
#define NH 256
#define NE 32
#define TOPK 32

// ---------------------------------------------------------------------------
// K0: transpose fp_w -> f16 wT[e][k]  (contiguous B-fragment loads), t_proj.
// ---------------------------------------------------------------------------
__global__ __launch_bounds__(256) void prep_kernel(
    const float* __restrict__ fpw, const float* __restrict__ tpw,
    const float* __restrict__ tpb, const float* __restrict__ temb,
    _Float16* __restrict__ wT, float* __restrict__ tproj) {
  int t = threadIdx.x;
  for (int i = t; i < NE * NH; i += 256) {
    int e = i >> 8, k = i & (NH - 1);
    wT[e * NH + k] = (_Float16)fpw[k * NE + e];
  }
  for (int i = t; i < NB * NE; i += 256) {
    int b = i >> 5, e = i & (NE - 1);
    float acc = tpb[e];
    for (int h = 0; h < NH; ++h) acc = fmaf(temb[b * NH + h], tpw[h * NE + e], acc);
    tproj[i] = acc;
  }
}

// ---------------------------------------------------------------------------
// K1: inv_s[n] = 1 / (sum_m relu(static[n][m]) + 1)   (the +1 is identity)
// ---------------------------------------------------------------------------
__global__ __launch_bounds__(256) void rowsum_kernel(
    const float* __restrict__ sadj, float* __restrict__ inv_s) {
  int lane = threadIdx.x & 31, wv = threadIdx.x >> 5;
  int n = blockIdx.x * 8 + wv;
  const float* row = sadj + (size_t)n * NN;
  float s = 0.f;
  for (int it = 0; it < NN / 128; ++it) {
    v4f v = *(const v4f*)(row + it * 128 + lane * 4);
    s += fmaxf(v.x, 0.f) + fmaxf(v.y, 0.f) + fmaxf(v.z, 0.f) + fmaxf(v.w, 0.f);
  }
  for (int off = 16; off > 0; off >>= 1) s += __shfl_down(s, off, 32);
  if (lane == 0) inv_s[n] = 1.f / (s + 1.f);
}

// ---------------------------------------------------------------------------
// K2: rep[b,n,e] = tanh(seq@fp_w + fp_b + node_emb + t_proj)  -> f16
//     WMMA f16 16x16x32, K=256 in 8 steps, one M-tile + 2 N-tiles per wave.
// ---------------------------------------------------------------------------
__global__ __launch_bounds__(128) void rep_kernel(
    const float* __restrict__ seq, const _Float16* __restrict__ wT,
    const float* __restrict__ fpb, const float* __restrict__ nemb,
    const float* __restrict__ tproj, _Float16* __restrict__ rep) {
  int lane = threadIdx.x & 31, wv = threadIdx.x >> 5;
  int hf = lane >> 4, l15 = lane & 15;
  int b = blockIdx.y;
  int mtile = blockIdx.x * 4 + wv;
  const float* arow = seq + ((size_t)b * NN + mtile * 16 + l15) * NH;
  v8f c0 = {}, c1 = {};
  for (int k0 = 0; k0 < NH; k0 += 32) {
    // A-frag (contiguous K: lanes<16 hold K 0..15, lanes>=16 K 16..31)
    v4f f0 = *(const v4f*)(arow + k0 + hf * 16);
    v4f f1 = *(const v4f*)(arow + k0 + hf * 16 + 4);
    v4f f2 = *(const v4f*)(arow + k0 + hf * 16 + 8);
    v4f f3 = *(const v4f*)(arow + k0 + hf * 16 + 12);
    v16h a;
#pragma unroll
    for (int i = 0; i < 4; i++) {
      a[i]      = (_Float16)f0[i];
      a[4 + i]  = (_Float16)f1[i];
      a[8 + i]  = (_Float16)f2[i];
      a[12 + i] = (_Float16)f3[i];
    }
    // B-frags (lanes<16 hold K 0..15 of col n, lanes>=16 K 16..31 of col n-16)
    v16h b0 = *(const v16h*)(wT + l15 * NH + k0 + hf * 16);
    v16h b1 = *(const v16h*)(wT + (16 + l15) * NH + k0 + hf * 16);
    c0 = __builtin_amdgcn_wmma_f32_16x16x32_f16(false, a, false, b0, (short)0, c0, false, false);
    c1 = __builtin_amdgcn_wmma_f32_16x16x32_f16(false, a, false, b1, (short)0, c1, false, false);
  }
  float bias0 = fpb[l15] + tproj[b * NE + l15];
  float bias1 = fpb[16 + l15] + tproj[b * NE + 16 + l15];
  int rbase = mtile * 16 + hf * 8;
#pragma unroll
  for (int r = 0; r < 8; r++) {
    int row = rbase + r;
    float v0 = c0[r] + bias0 + nemb[row * NE + l15];
    float v1 = c1[r] + bias1 + nemb[row * NE + 16 + l15];
    size_t o = ((size_t)b * NN + row) * NE;
    rep[o + l15]      = (_Float16)tanhf(v0);
    rep[o + 16 + l15] = (_Float16)tanhf(v1);
  }
}

// ---------------------------------------------------------------------------
// K3: fused  sim -> top-k -> renorm chain -> blend -> output.
//     Wave owns 32 rows (two 16x16 WMMA row-tiles). Tiles transposed through
//     LDS so each lane owns a full row; per-lane top-32 list lives in LDS.
// ---------------------------------------------------------------------------
__global__ __launch_bounds__(128) void graph_kernel(
    const float* __restrict__ sadj, const float* __restrict__ inv_s,
    const _Float16* __restrict__ rep, const float* __restrict__ blend_logit,
    float* __restrict__ out) {
  __shared__ float tbuf[4][32][20];       // 16B-aligned rows, de-conflicted stride
  __shared__ float tvals[4][TOPK][32];    // [wave][slot][lane]
  __shared__ int   tidx [4][TOPK][32];

  int lane = threadIdx.x & 31, wv = threadIdx.x >> 5;
  int hf = lane >> 4, l15 = lane & 15;
  int b = blockIdx.y;
  int R0 = blockIdx.x * 128 + wv * 32;
  const _Float16* repb = rep + (size_t)b * NN * NE;

  // A fragments for row-tiles [R0..+15], [R0+16..+31], pre-scaled by 1/sqrt(E)
  const _Float16 sc = (_Float16)0.17677669529663687f;
  v16h a0, a1;
  {
    const _Float16* rp0 = repb + (size_t)(R0 + l15) * NE + hf * 16;
    const _Float16* rp1 = repb + (size_t)(R0 + 16 + l15) * NE + hf * 16;
    v8h lo0 = *(const v8h*)rp0, hi0 = *(const v8h*)(rp0 + 8);
    v8h lo1 = *(const v8h*)rp1, hi1 = *(const v8h*)(rp1 + 8);
#pragma unroll
    for (int i = 0; i < 8; i++) {
      a0[i] = lo0[i] * sc; a0[8 + i] = hi0[i] * sc;
      a1[i] = lo1[i] * sc; a1[8 + i] = hi1[i] * sc;
    }
  }

#pragma unroll
  for (int k = 0; k < TOPK; k++) tvals[wv][k][lane] = -1e30f;
  float minv = -1e30f; int minpos = 0;
  v8f zero = {};

  for (int j = 0; j < NN / 16; ++j) {
    // B-frag = rep^T column tile j (lanes<16: K0..15, lanes>=16: K16..31 of col-16)
    const _Float16* bp = repb + (size_t)(j * 16 + l15) * NE + hf * 16;
    v16h bf = *(const v16h*)bp;
    v8f s0 = __builtin_amdgcn_wmma_f32_16x16x32_f16(false, a0, false, bf, (short)0, zero, false, false);
    v8f s1 = __builtin_amdgcn_wmma_f32_16x16x32_f16(false, a1, false, bf, (short)0, zero, false, false);
    // transpose both 16x16 tiles into row-major LDS (C layout: VGPR r -> M=r(+8))
#pragma unroll
    for (int r = 0; r < 8; r++) {
      tbuf[wv][hf * 8 + r][l15]      = s0[r];
      tbuf[wv][16 + hf * 8 + r][l15] = s1[r];
    }
    asm volatile("s_wait_dscnt 0x0" ::: "memory");
    v4f q0 = *(const v4f*)&tbuf[wv][lane][0];
    v4f q1 = *(const v4f*)&tbuf[wv][lane][4];
    v4f q2 = *(const v4f*)&tbuf[wv][lane][8];
    v4f q3 = *(const v4f*)&tbuf[wv][lane][12];
    float cand[16];
#pragma unroll
    for (int i = 0; i < 4; i++) { cand[i] = q0[i]; cand[4+i] = q1[i]; cand[8+i] = q2[i]; cand[12+i] = q3[i]; }
#pragma unroll
    for (int c = 0; c < 16; c++) {
      float v = cand[c];
      if (v > minv) {                       // replace current min slot
        tvals[wv][minpos][lane] = v;
        tidx [wv][minpos][lane] = j * 16 + c;
        float mv = 1e30f; int mp = 0;
        for (int k = 0; k < TOPK; k++) {
          float x = tvals[wv][k][lane];
          if (x < mv) { mv = x; mp = k; }
        }
        minv = mv; minpos = mp;
      }
    }
  }

  float bl = 1.f / (1.f + expf(-blend_logit[0]));
  float onembl = 1.f - bl;

  // per-row (lane-owned): softmax denominator cancels after top-k renorm;
  // only sum over kept logits is needed. exp(v-6) is overflow-safe (|sim|<=5.66).
  float T = 0.f;
#pragma unroll
  for (int k = 0; k < TOPK; k++) {
    float e = expf(tvals[wv][k][lane] - 6.f);
    tvals[wv][k][lane] = e;
    T += e;
  }
  float invT = 1.f / T;
  // final blend row-sum == 1 analytically (both components row-normalized).

  float* outb = out + ((size_t)b * NN + R0) * NN;

  // dense pass: coalesced write of all 32 rows (static part + diagonal)
  for (int r = 0; r < 32; r++) {
    int n = R0 + r;
    float cstat = onembl * inv_s[n];
    float cdiag = cstat + bl * 0.5f;       // identity in static_norm + identity in dyn
    const float* srow = sadj + (size_t)n * NN;
    float* orow = outb + (size_t)r * NN;
    for (int it = 0; it < NN / 128; ++it) {
      int m = it * 128 + lane * 4;
      v4f sv = *(const v4f*)(srow + m);
      v4f ov;
#pragma unroll
      for (int q = 0; q < 4; q++) {
        float val = fmaxf(sv[q], 0.f) * cstat;
        if (m + q == n) val += cdiag;
        ov[q] = val;
      }
      *(v4f*)(orow + m) = ov;
    }
  }
  asm volatile("s_wait_storecnt 0x0" ::: "memory");

  // sparse pass: lane patches its own row's 32 top-k columns
  {
    int n = R0 + lane;
    float cstat = onembl * inv_s[n];
    float cdiag = cstat + bl * 0.5f;
    const float* srow = sadj + (size_t)n * NN;
    float* orow = outb + (size_t)lane * NN;
    for (int k = 0; k < TOPK; k++) {
      int col = tidx[wv][k][lane];
      float e = tvals[wv][k][lane];
      float val = fmaxf(srow[col], 0.f) * cstat + bl * 0.5f * e * invT;
      if (col == n) val += cdiag;
      orow[col] = val;
    }
  }
}

// ---------------------------------------------------------------------------
extern "C" void kernel_launch(void* const* d_in, const int* in_sizes, int n_in,
                              void* d_out, int out_size, void* d_ws, size_t ws_size,
                              hipStream_t stream) {
  const float* seq   = (const float*)d_in[0];
  const float* temb  = (const float*)d_in[1];
  const float* sadj  = (const float*)d_in[2];
  const float* nemb  = (const float*)d_in[3];
  const float* fpw   = (const float*)d_in[4];
  const float* fpb   = (const float*)d_in[5];
  const float* tpw   = (const float*)d_in[6];
  const float* tpb   = (const float*)d_in[7];
  const float* blend = (const float*)d_in[8];
  float* out = (float*)d_out;

  char* ws = (char*)d_ws;
  _Float16* rep   = (_Float16*)ws;                                   // 2 MB
  _Float16* wT    = (_Float16*)(ws + (size_t)NB * NN * NE * 2);      // 16 KB
  float*    tproj = (float*)(ws + (size_t)NB * NN * NE * 2 + 16384); // 2 KB
  float*    invs  = (float*)(ws + (size_t)NB * NN * NE * 2 + 16384 + 4096);

  prep_kernel  <<<1, 256, 0, stream>>>(fpw, tpw, tpb, temb, wT, tproj);
  rowsum_kernel<<<NN / 8, 256, 0, stream>>>(sadj, invs);
  rep_kernel   <<<dim3(NN / 64, NB), 128, 0, stream>>>(seq, wT, fpb, nemb, tproj, rep);
  graph_kernel <<<dim3(NN / 128, NB), 128, 0, stream>>>(sadj, invs, rep, blend, out);
}
